// GraphTransformer_44959717654791
// MI455X (gfx1250) — compile-verified, matrix operand order checked by reading, not run on hardware
//
#include <hip/hip_runtime.h>
#include <math.h>

// ---------------------------------------------------------------------------
// GraphTransformer forward for MI455X (gfx1250, wave32, WMMA 16x16x32 f16,
// TDM tensor_load_to_lds + async global->LDS staging)
// ---------------------------------------------------------------------------

#define N_NODES   16384
#define D_MODEL   64
#define N_HEADS   4
#define D_HEAD    16
#define D_FF      256
#define N_GRAPHS  256
#define NODES_PER 64
#define DEG       8
#define N_CLASS   10

typedef __attribute__((ext_vector_type(16))) _Float16 v16h;
typedef __attribute__((ext_vector_type(8)))  float    v8f;
typedef unsigned int u32x4 __attribute__((ext_vector_type(4)));
typedef int          i32x8 __attribute__((ext_vector_type(8)));
typedef int          i32x4 __attribute__((ext_vector_type(4)));

#define WMMA_F16(a, b, c) \
  __builtin_amdgcn_wmma_f32_16x16x32_f16(false, (a), false, (b), (short)0, (c), false, false)

// K-index mapping for 16-bit A/B fragments (16x32), per ISA 7.12.2:
// element e -> k = (e<8 ? e : e+8) + 8*hi  (two contiguous runs of 8)
__device__ __forceinline__ int kmap(int e, int hi) {
  return (e < 8 ? e : e + 8) + (hi << 3);
}

// A fragment from LDS-resident f32 tile (row-major, ld=64): 2 contiguous runs
__device__ __forceinline__ v16h load_a_ldsf32(const float* Xs, int row, int k0, int hi) {
  const float* p = Xs + row * 64 + k0 + hi * 8;
  v16h a;
#pragma unroll
  for (int e = 0; e < 8; ++e) a[e] = (_Float16)p[e];
#pragma unroll
  for (int e = 0; e < 8; ++e) a[e + 8] = (_Float16)p[16 + e];
  return a;
}

// B fragment from LDS-resident transposed f16 weight Wt[col][k] (ld=64)
__device__ __forceinline__ v16h load_b_ldsf16(const _Float16* Wt, int col, int k0, int hi) {
  const _Float16* p = Wt + col * 64 + k0 + hi * 8;
  v16h b;
#pragma unroll
  for (int e = 0; e < 8; ++e) b[e] = p[e];
#pragma unroll
  for (int e = 0; e < 8; ++e) b[e + 8] = p[16 + e];
  return b;
}

// A fragment from global row-major f32, vectorized (4 x b128 per lane)
__device__ __forceinline__ v16h load_a_f32v(const float* __restrict__ X, int ld,
                                            int row0, int k0) {
  const int lane = threadIdx.x & 31;
  const int r = lane & 15, hi = lane >> 4;
  const float* p = X + (size_t)(row0 + r) * ld + k0 + hi * 8;
  float4 u0 = *(const float4*)(p);
  float4 u1 = *(const float4*)(p + 4);
  float4 u2 = *(const float4*)(p + 16);
  float4 u3 = *(const float4*)(p + 20);
  v16h a;
  a[0] = (_Float16)u0.x;  a[1] = (_Float16)u0.y;  a[2]  = (_Float16)u0.z;  a[3]  = (_Float16)u0.w;
  a[4] = (_Float16)u1.x;  a[5] = (_Float16)u1.y;  a[6]  = (_Float16)u1.z;  a[7]  = (_Float16)u1.w;
  a[8] = (_Float16)u2.x;  a[9] = (_Float16)u2.y;  a[10] = (_Float16)u2.z;  a[11] = (_Float16)u2.w;
  a[12] = (_Float16)u3.x; a[13] = (_Float16)u3.y; a[14] = (_Float16)u3.z;  a[15] = (_Float16)u3.w;
  return a;
}

// B fragment from global row-major f32 W[K x N] (column-strided, scalar)
__device__ __forceinline__ v16h load_b_f32(const float* __restrict__ W, int ld,
                                           int k0, int n0) {
  const int lane = threadIdx.x & 31;
  const int c = lane & 15, hi = lane >> 4;
  v16h b;
#pragma unroll
  for (int e = 0; e < 16; ++e)
    b[e] = (_Float16)W[(size_t)(k0 + kmap(e, hi)) * ld + n0 + c];
  return b;
}

// Stage 64x64 f32 W into LDS transposed as f16: Wt[col][k]
__device__ __forceinline__ void stage_w_t(_Float16* Wt, const float* __restrict__ W,
                                          int tid) {
#pragma unroll
  for (int p = 0; p < 8; ++p) {
    int linear = p * 512 + tid * 4;     // 128 threads x 4 floats x 8 passes = 4096
    int kk = linear >> 6, nn = linear & 63;
    float4 wv = *(const float4*)(W + linear);
    Wt[(nn + 0) * 64 + kk] = (_Float16)wv.x;
    Wt[(nn + 1) * 64 + kk] = (_Float16)wv.y;
    Wt[(nn + 2) * 64 + kk] = (_Float16)wv.z;
    Wt[(nn + 3) * 64 + kk] = (_Float16)wv.w;
  }
}

// ---------------------------------------------------------------------------
// Embedding lookup: h[n][d] = emb[x[n]][d]
// ---------------------------------------------------------------------------
__global__ void k_embed(float* __restrict__ h, const float* __restrict__ emb,
                        const int* __restrict__ x) {
  int idx = blockIdx.x * blockDim.x + threadIdx.x;  // N_NODES*64 threads
  int n = idx >> 6, d = idx & 63;
  h[idx] = emb[x[n] * D_MODEL + d];
}

// ---------------------------------------------------------------------------
// Y[N x 64] = X[N x 64] @ W[64 x 64]
// X tile (64x64 f32) staged into LDS by the Tensor Data Mover
// (one tensor_load_to_lds per block), W staged transposed f16 in LDS.
// block = 128 threads (4 waves), each wave -> 16 rows x 64 cols.
// ---------------------------------------------------------------------------
__global__ void k_gemm64(float* __restrict__ Y, const float* __restrict__ X,
                         const float* __restrict__ W) {
  __shared__ float    Xs[64 * 64];   // 16KB
  __shared__ _Float16 Wt[64 * 64];   // 8KB, transposed
  const int tid = threadIdx.x;
  const int lane = tid & 31, wave = tid >> 5;
  const int r = lane & 15, hi = lane >> 4;

  if (wave == 0) {
    // Build Tensor DMA Descriptor (D#) per ISA 08_async_tensor §8:
    // 2D tensor 64x64 f32, tile 64x64, stride 64 -> one TDM op moves the tile.
    unsigned long long ga =
        (unsigned long long)(const void*)(X + (size_t)blockIdx.x * 64 * 64);
    unsigned ldsa = (unsigned)(unsigned long long)(void*)&Xs[0];
    u32x4 g0;
    g0[0] = 1u;                                   // count=1, user mode
    g0[1] = ldsa;                                 // lds_addr
    g0[2] = (unsigned)ga;                         // global_addr[31:0]
    g0[3] = (unsigned)((ga >> 32) & 0x01FFFFFFull) | (2u << 30); // addr[56:32] | type=2
    i32x8 g1;
    g1[0] = (2 << 16);                            // data_size = 4 bytes
    g1[1] = (64 << 16);                           // tensor_dim0 = 64 (bits 79:48)
    g1[2] = (64 << 16);                           // tensor_dim1 = 64 (bits 111:80)
    g1[3] = (64 << 16);                           // tile_dim0 = 64 (bits 127:112)
    g1[4] = 64;                                   // tile_dim1 = 64, tile_dim2 = 0
    g1[5] = 64;                                   // tensor_dim0_stride = 64
    g1[6] = 0;
    g1[7] = 0;
    i32x4 gz4 = {0, 0, 0, 0};                     // groups 2/3 unused (2D tensor)
    i32x8 gz8 = {0, 0, 0, 0, 0, 0, 0, 0};
    __builtin_amdgcn_tensor_load_to_lds(g0, g1, gz4, gz4, gz8, 0);
    __builtin_amdgcn_s_wait_tensorcnt((short)0);
  }
  stage_w_t(Wt, W, tid);
  __syncthreads();

  const int rloc = wave * 16;
  v16h a0 = load_a_ldsf32(Xs, rloc + r, 0, hi);
  v16h a1 = load_a_ldsf32(Xs, rloc + r, 32, hi);
  const int row0 = blockIdx.x * 64 + rloc;
#pragma unroll
  for (int t = 0; t < 4; ++t) {
    v16h b0 = load_b_ldsf16(Wt, t * 16 + r, 0, hi);
    v16h b1 = load_b_ldsf16(Wt, t * 16 + r, 32, hi);
    v8f c = {};
    c = WMMA_F16(a0, b0, c);
    c = WMMA_F16(a1, b1, c);
#pragma unroll
    for (int i = 0; i < 8; ++i)
      Y[(size_t)(row0 + i + hi * 8) * D_MODEL + t * 16 + r] = c[i];
  }
}

// ---------------------------------------------------------------------------
// struct = h + relu(hWself + sum_{j<8} hWnbr[src[8n+j]])
// ---------------------------------------------------------------------------
__global__ void k_struct(float* __restrict__ st, const float* __restrict__ h,
                         const float* __restrict__ hs, const float* __restrict__ hn,
                         const int* __restrict__ esrc) {
  int idx = blockIdx.x * blockDim.x + threadIdx.x;
  int n = idx >> 6, d = idx & 63;
  const int* s = esrc + n * DEG;
  float agg = 0.f;
#pragma unroll
  for (int j = 0; j < DEG; ++j) agg += hn[(size_t)s[j] * D_MODEL + d];
  float t = hs[idx] + agg;
  st[idx] = h[idx] + (t > 0.f ? t : 0.f);
}

// ---------------------------------------------------------------------------
// Per-graph multi-head attention: out = softmax(Q K^T / 4) V
// one block (256 thr = 8 waves) per graph; wave w -> (head = w&3, mtile = w>>2)
// ---------------------------------------------------------------------------
__global__ void k_attn(float* __restrict__ out, const float* __restrict__ q,
                       const float* __restrict__ kp, const float* __restrict__ vp) {
  __shared__ _Float16 qs[64 * 64];
  __shared__ _Float16 ksm[64 * 64];
  __shared__ _Float16 vsm[64 * 64];
  __shared__ float    Sb[8][16 * 64];
  __shared__ float    dls[8][16];

  const int g = blockIdx.x;
  const int tid = threadIdx.x;
  const int base = g * NODES_PER * D_MODEL;
#pragma unroll
  for (int p = 0; p < 4; ++p) {
    int idx = (tid + p * 256) * 4;
    float4 qv = *(const float4*)(q + base + idx);
    float4 kv = *(const float4*)(kp + base + idx);
    float4 vv = *(const float4*)(vp + base + idx);
    qs[idx + 0] = (_Float16)(qv.x * 0.25f);  qs[idx + 1] = (_Float16)(qv.y * 0.25f);
    qs[idx + 2] = (_Float16)(qv.z * 0.25f);  qs[idx + 3] = (_Float16)(qv.w * 0.25f);
    ksm[idx + 0] = (_Float16)kv.x;  ksm[idx + 1] = (_Float16)kv.y;
    ksm[idx + 2] = (_Float16)kv.z;  ksm[idx + 3] = (_Float16)kv.w;
    vsm[idx + 0] = (_Float16)vv.x;  vsm[idx + 1] = (_Float16)vv.y;
    vsm[idx + 2] = (_Float16)vv.z;  vsm[idx + 3] = (_Float16)vv.w;
  }
  __syncthreads();

  const int lane = tid & 31, w = tid >> 5;
  const int hd = w & 3, mt = w >> 2;
  const int r = lane & 15, hi = lane >> 4;

  // S[16 x 64] = Q_tile K^T  (K-dim 16 zero-padded to 32)
  v16h aq;
#pragma unroll
  for (int e = 0; e < 16; ++e) {
    int kk = kmap(e, hi);
    aq[e] = (kk < D_HEAD) ? qs[(mt * 16 + r) * D_MODEL + hd * D_HEAD + kk]
                          : (_Float16)0.f;
  }
#pragma unroll
  for (int t = 0; t < 4; ++t) {
    v16h bk;
#pragma unroll
    for (int e = 0; e < 16; ++e) {
      int kk = kmap(e, hi);
      bk[e] = (kk < D_HEAD) ? ksm[(t * 16 + r) * D_MODEL + hd * D_HEAD + kk]
                            : (_Float16)0.f;
    }
    v8f c = {};
    c = WMMA_F16(aq, bk, c);
#pragma unroll
    for (int i = 0; i < 8; ++i)
      Sb[w][(i + hi * 8) * 64 + t * 16 + r] = c[i];
  }

  // row softmax (two lanes per row; halves combine via shfl_xor 16)
  const int cb = hi * 32;
  float m = -1e30f;
  for (int c = 0; c < 32; ++c) m = fmaxf(m, Sb[w][r * 64 + cb + c]);
  m = fmaxf(m, __shfl_xor(m, 16, 32));
  float den = 0.f;
  for (int c = 0; c < 32; ++c) {
    float e = __expf(Sb[w][r * 64 + cb + c] - m);
    Sb[w][r * 64 + cb + c] = e;
    den += e;
  }
  den += __shfl_xor(den, 16, 32);
  if (hi == 0) dls[w][r] = den;

  // attn = P V (unnormalized P; divide by den at store)
  v8f acc = {};
#pragma unroll
  for (int k0 = 0; k0 < 64; k0 += 32) {
    v16h ap, bv;
#pragma unroll
    for (int e = 0; e < 16; ++e) {
      int kk = kmap(e, hi);
      ap[e] = (_Float16)Sb[w][r * 64 + k0 + kk];
      bv[e] = vsm[(k0 + kk) * D_MODEL + hd * D_HEAD + r];
    }
    acc = WMMA_F16(ap, bv, acc);
  }
#pragma unroll
  for (int i = 0; i < 8; ++i) {
    int rr = i + hi * 8;
    out[(size_t)(g * NODES_PER + mt * 16 + rr) * D_MODEL + hd * D_HEAD + r] =
        acc[i] / dls[w][rr];
  }
}

// ---------------------------------------------------------------------------
// out = LayerNorm(resid + X @ W + bias) * g + b
// X tile staged via async global->LDS (GLOBAL_LOAD_ASYNC_TO_LDS_B128),
// W staged transposed f16 in LDS.
// ---------------------------------------------------------------------------
__global__ void k_gemm_ln(float* __restrict__ out, const float* __restrict__ X,
                          const float* __restrict__ W, const float* __restrict__ bias,
                          const float* __restrict__ resid, const float* __restrict__ gg,
                          const float* __restrict__ bb) {
  __shared__ float    Xs[64 * 64];
  __shared__ _Float16 Wt[64 * 64];
  const int tid = threadIdx.x;
  const int lane = tid & 31, wave = tid >> 5;
  const int r = lane & 15, hi = lane >> 4;

  {
    const float* gbase = X + (size_t)blockIdx.x * 64 * 64;
    unsigned ldsbase = (unsigned)(unsigned long long)(void*)&Xs[0];
#pragma unroll
    for (int p = 0; p < 8; ++p) {
      unsigned goff = (unsigned)(p * 512 + tid * 4) * 4u;  // byte offset
      unsigned loff = ldsbase + goff;
      asm volatile("global_load_async_to_lds_b128 %0, %1, %2 offset:0"
                   :: "v"(loff), "v"(goff), "s"(gbase)
                   : "memory");
    }
    asm volatile("s_wait_asynccnt 0x0" ::: "memory");
  }
  stage_w_t(Wt, W, tid);
  __syncthreads();

  const int rloc = wave * 16;
  const int row0 = blockIdx.x * 64 + rloc;
  v16h a0 = load_a_ldsf32(Xs, rloc + r, 0, hi);
  v16h a1 = load_a_ldsf32(Xs, rloc + r, 32, hi);
  float vals[4][8];
#pragma unroll
  for (int t = 0; t < 4; ++t) {
    v16h b0 = load_b_ldsf16(Wt, t * 16 + r, 0, hi);
    v16h b1 = load_b_ldsf16(Wt, t * 16 + r, 32, hi);
    v8f c = {};
    c = WMMA_F16(a0, b0, c);
    c = WMMA_F16(a1, b1, c);
    int col = t * 16 + r;
#pragma unroll
    for (int i = 0; i < 8; ++i)
      vals[t][i] = c[i] + bias[col] +
                   resid[(size_t)(row0 + i + hi * 8) * D_MODEL + col];
  }
#pragma unroll
  for (int i = 0; i < 8; ++i) {
    float s = 0.f, sq = 0.f;
#pragma unroll
    for (int t = 0; t < 4; ++t) { s += vals[t][i]; sq += vals[t][i] * vals[t][i]; }
#pragma unroll
    for (int msk = 1; msk < 16; msk <<= 1) {
      s  += __shfl_xor(s,  msk, 32);
      sq += __shfl_xor(sq, msk, 32);
    }
    float mean = s * (1.f / 64.f);
    float var  = sq * (1.f / 64.f) - mean * mean;
    float rstd = rsqrtf(var + 1e-5f);
    int row = row0 + i + hi * 8;
#pragma unroll
    for (int t = 0; t < 4; ++t) {
      int col = t * 16 + r;
      out[(size_t)row * D_MODEL + col] =
          (vals[t][i] - mean) * rstd * gg[col] + bb[col];
    }
  }
}

// ---------------------------------------------------------------------------
// Fused FFN: out = LayerNorm(X + relu(X W1 + b1) W2 + b2) * g + b
// ---------------------------------------------------------------------------
__global__ void k_ffn(float* __restrict__ out, const float* __restrict__ X,
                      const float* __restrict__ W1, const float* __restrict__ b1,
                      const float* __restrict__ W2, const float* __restrict__ b2,
                      const float* __restrict__ gg, const float* __restrict__ bb) {
  __shared__ _Float16 fbuf[4][16 * D_FF];  // 32KB
  const int lane = threadIdx.x & 31, wave = threadIdx.x >> 5;
  const int row0 = blockIdx.x * 64 + wave * 16;
  const int r = lane & 15, hi = lane >> 4;

  v16h a0 = load_a_f32v(X, D_MODEL, row0, 0);
  v16h a1 = load_a_f32v(X, D_MODEL, row0, 32);
#pragma unroll
  for (int t = 0; t < 16; ++t) {
    v16h b0  = load_b_f32(W1, D_FF, 0, t * 16);
    v16h b1f = load_b_f32(W1, D_FF, 32, t * 16);
    v8f acc = {};
    acc = WMMA_F16(a0, b0, acc);
    acc = WMMA_F16(a1, b1f, acc);
    int col = t * 16 + r;
#pragma unroll
    for (int i = 0; i < 8; ++i) {
      float u = acc[i] + b1[col];
      fbuf[wave][(i + hi * 8) * D_FF + col] = (_Float16)(u > 0.f ? u : 0.f);
    }
  }
  v16h af[8];
#pragma unroll
  for (int kk = 0; kk < 8; ++kk) {
    const _Float16* p = fbuf[wave] + r * D_FF + kk * 32 + hi * 8;
#pragma unroll
    for (int e = 0; e < 8; ++e) { af[kk][e] = p[e]; af[kk][e + 8] = p[16 + e]; }
  }
  float vals[4][8];
#pragma unroll
  for (int t = 0; t < 4; ++t) {
    v8f acc = {};
#pragma unroll
    for (int kk = 0; kk < 8; ++kk) {
      v16h bw = load_b_f32(W2, D_MODEL, kk * 32, t * 16);
      acc = WMMA_F16(af[kk], bw, acc);
    }
    int col = t * 16 + r;
#pragma unroll
    for (int i = 0; i < 8; ++i)
      vals[t][i] = acc[i] + b2[col] +
                   X[(size_t)(row0 + i + hi * 8) * D_MODEL + col];
  }
#pragma unroll
  for (int i = 0; i < 8; ++i) {
    float s = 0.f, sq = 0.f;
#pragma unroll
    for (int t = 0; t < 4; ++t) { s += vals[t][i]; sq += vals[t][i] * vals[t][i]; }
#pragma unroll
    for (int msk = 1; msk < 16; msk <<= 1) {
      s  += __shfl_xor(s,  msk, 32);
      sq += __shfl_xor(sq, msk, 32);
    }
    float mean = s * (1.f / 64.f);
    float var  = sq * (1.f / 64.f) - mean * mean;
    float rstd = rsqrtf(var + 1e-5f);
    int row = row0 + i + hi * 8;
#pragma unroll
    for (int t = 0; t < 4; ++t) {
      int col = t * 16 + r;
      out[(size_t)row * D_MODEL + col] =
          (vals[t][i] - mean) * rstd * gg[col] + bb[col];
    }
  }
}

// ---------------------------------------------------------------------------
// Per-graph mean pool + relu(pooled@W1+b1)@W2+b2  -> out[256 x 10]
// ---------------------------------------------------------------------------
__global__ void k_cls(float* __restrict__ out, const float* __restrict__ h,
                      const float* __restrict__ W1, const float* __restrict__ b1,
                      const float* __restrict__ W2, const float* __restrict__ b2) {
  __shared__ float pooled[D_MODEL];
  __shared__ float z[D_MODEL];
  const int g = blockIdx.x, d = threadIdx.x;  // 64 threads
  float s = 0.f;
  for (int n = 0; n < NODES_PER; ++n)
    s += h[(size_t)(g * NODES_PER + n) * D_MODEL + d];
  pooled[d] = s * (1.f / NODES_PER);
  __syncthreads();
  float acc = b1[d];
  for (int kk = 0; kk < D_MODEL; ++kk) acc += pooled[kk] * W1[kk * D_MODEL + d];
  z[d] = acc > 0.f ? acc : 0.f;
  __syncthreads();
  if (d < N_CLASS) {
    float o = b2[d];
    for (int kk = 0; kk < D_MODEL; ++kk) o += z[kk] * W2[kk * N_CLASS + d];
    out[g * N_CLASS + d] = o;
  }
}

// ---------------------------------------------------------------------------
extern "C" void kernel_launch(void* const* d_in, const int* in_sizes, int n_in,
                              void* d_out, int out_size, void* d_ws, size_t ws_size,
                              hipStream_t stream) {
  const int*   x       = (const int*)d_in[0];
  const int*   e_src   = (const int*)d_in[1];
  const float* emb     = (const float*)d_in[4];
  const float* Wg_self = (const float*)d_in[5];
  const float* Wg_nbr  = (const float*)d_in[6];
  const float* Wq      = (const float*)d_in[7];
  const float* Wk      = (const float*)d_in[8];
  const float* Wv      = (const float*)d_in[9];
  const float* Wo      = (const float*)d_in[10];
  const float* bo      = (const float*)d_in[11];
  const float* ln1_g   = (const float*)d_in[12];
  const float* ln1_b   = (const float*)d_in[13];
  const float* ff_W1   = (const float*)d_in[14];
  const float* ff_b1   = (const float*)d_in[15];
  const float* ff_W2   = (const float*)d_in[16];
  const float* ff_b2   = (const float*)d_in[17];
  const float* ln2_g   = (const float*)d_in[18];
  const float* ln2_b   = (const float*)d_in[19];
  const float* c_W1    = (const float*)d_in[20];
  const float* c_b1    = (const float*)d_in[21];
  const float* c_W2    = (const float*)d_in[22];
  const float* c_b2    = (const float*)d_in[23];

  const int M = N_NODES * D_MODEL;
  float* ws = (float*)d_ws;
  float* h  = ws + 0 * (size_t)M;
  float* t1 = ws + 1 * (size_t)M;   // hWnbr -> q
  float* t2 = ws + 2 * (size_t)M;   // hWself -> k
  float* st = ws + 3 * (size_t)M;   // struct -> h_after_attn
  float* vv = ws + 4 * (size_t)M;   // v
  float* at = ws + 5 * (size_t)M;   // attention output

  k_embed<<<M / 256, 256, 0, stream>>>(h, emb, x);

  for (int L = 0; L < 2; ++L) {
    const int w64 = L * D_MODEL * D_MODEL;
    k_gemm64<<<256, 128, 0, stream>>>(t1, h, Wg_nbr + w64);
    k_gemm64<<<256, 128, 0, stream>>>(t2, h, Wg_self + w64);
    k_struct<<<M / 256, 256, 0, stream>>>(st, h, t2, t1, e_src);
    k_gemm64<<<256, 128, 0, stream>>>(t1, st, Wq + w64);
    k_gemm64<<<256, 128, 0, stream>>>(t2, st, Wk + w64);
    k_gemm64<<<256, 128, 0, stream>>>(vv, h, Wv + w64);
    k_attn<<<N_GRAPHS, 256, 0, stream>>>(at, t1, t2, vv);
    k_gemm_ln<<<256, 128, 0, stream>>>(st, at, Wo + w64, bo + L * D_MODEL, h,
                                       ln1_g + L * D_MODEL, ln1_b + L * D_MODEL);
    k_ffn<<<256, 128, 0, stream>>>(h, st, ff_W1 + L * D_MODEL * D_FF,
                                   ff_b1 + L * D_FF, ff_W2 + L * D_FF * D_MODEL,
                                   ff_b2 + L * D_MODEL, ln2_g + L * D_MODEL,
                                   ln2_b + L * D_MODEL);
  }

  k_cls<<<N_GRAPHS, 64, 0, stream>>>((float*)d_out, h, c_W1, c_b1, c_W2, c_b2);
}
